// QuantumGraphConv_26018911879287
// MI455X (gfx1250) — compile-verified, matrix operand order checked by reading, not run on hardware
//
#include <hip/hip_runtime.h>
#include <math.h>

// ---------------------------------------------------------------------------
// CDNA5 (gfx1250) implementation: bf16 WMMA GEMMs (double-buffered LDS) +
// wave32 8-qubit state-vector simulator.
// ---------------------------------------------------------------------------

typedef __attribute__((ext_vector_type(16))) __bf16 v16bf;
typedef __attribute__((ext_vector_type(8)))  __bf16 v8bf;
typedef __attribute__((ext_vector_type(8)))  float  v8f;

// ---------------- f32 -> bf16 conversion ----------------
__global__ void cvt_f32_bf16_kernel(const float* __restrict__ in,
                                    __bf16* __restrict__ out, int n) {
  int i = blockIdx.x * blockDim.x + threadIdx.x;
  if (i < n) out[i] = (__bf16)in[i];
}

// ---------------- tiled transpose + f32 -> bf16 ----------------
// in:  (rows x cols) f32 row-major (per batch)
// out: (cols x rows) bf16 row-major (per batch)
__global__ __launch_bounds__(256)
void transpose_cvt_kernel(const float* __restrict__ in, __bf16* __restrict__ outT,
                          int rows, int cols) {
  __shared__ float tile[32][33];
  const int b = blockIdx.z;
  in   += (size_t)b * rows * cols;
  outT += (size_t)b * rows * cols;
  const int c0 = blockIdx.x * 32;
  const int r0 = blockIdx.y * 32;
  const int tx = threadIdx.x;        // 0..31
  const int ty = threadIdx.y;        // 0..7
  #pragma unroll
  for (int i = 0; i < 32; i += 8)
    tile[ty + i][tx] = in[(size_t)(r0 + ty + i) * cols + (c0 + tx)];
  __syncthreads();
  #pragma unroll
  for (int i = 0; i < 32; i += 8)
    outT[(size_t)(c0 + ty + i) * rows + (r0 + tx)] = (__bf16)tile[tx][ty + i];
}

// ---------------- WMMA GEMM:  C[M,N] = A[M,K] * Bt[N,K]^T ----------------
// EPI=0: store bf16, no bias. EPI=1: +bias then exact GELU, store f32.
// Block: 128 threads (4 waves). Block tile 64x64, K-step 32, LDS double buffer.
// Each wave owns a 32x32 quadrant: 2 A-frags x 2 B-frags -> 4 WMMAs/K-step.
__device__ __forceinline__ v16bf ldfrag(const __bf16* p) {
  // ISA 16-bit A/B layout: per-lane 16 elements as two contiguous 8-element
  // halves (K {0..7,16..23} for lanes 0-15; {8..15,24..31} for lanes 16-31).
  v8bf lo = *(const v8bf*)p;
  v8bf hi = *(const v8bf*)(p + 16);
  v16bf r;
  #pragma unroll
  for (int i = 0; i < 8; ++i) { r[i] = lo[i]; r[i + 8] = hi[i]; }
  return r;
}

template<int EPI>
__global__ __launch_bounds__(128)
void gemm_wmma_bf16(const __bf16* __restrict__ A, size_t aBatch,
                    const __bf16* __restrict__ Bt, size_t bBatch,
                    void* __restrict__ C, size_t cBatch,
                    const float* __restrict__ bias,
                    int K, int lda, int ldb, int ldc) {
  __shared__ __align__(16) __bf16 As[2][64][32];   // As[buf][m][k]
  __shared__ __align__(16) __bf16 Bs[2][64][32];   // Bs[buf][n][k]

  const int b = blockIdx.z;
  A  += aBatch * (size_t)b;
  Bt += bBatch * (size_t)b;

  const int n0blk = blockIdx.x * 64;
  const int m0blk = blockIdx.y * 64;
  const int tid   = threadIdx.x;
  const int lane  = tid & 31;
  const int wave  = tid >> 5;
  const int wm    = (wave & 1) * 32;    // wave row offset in block tile
  const int wn    = (wave >> 1) * 32;   // wave col offset in block tile
  const int mlane = lane & 15;
  const int khalf = (lane >= 16) ? 8 : 0;

  // Staging: 64x32 tile each for A and Bt; 128 threads x 16 contiguous elems.
  const int smm = tid >> 1;             // 0..63
  const int skk = (tid & 1) * 16;       // 0 or 16
  const __bf16* aPtr = A  + (size_t)(m0blk + smm) * lda + skk;
  const __bf16* bPtr = Bt + (size_t)(n0blk + smm) * ldb + skk;

  v8bf ra0, ra1, rb0, rb1;
  const int nIter = K / 32;

  // prologue: global -> regs -> LDS buf 0
  ra0 = *(const v8bf*)(aPtr);      ra1 = *(const v8bf*)(aPtr + 8);
  rb0 = *(const v8bf*)(bPtr);      rb1 = *(const v8bf*)(bPtr + 8);
  *(v8bf*)&As[0][smm][skk] = ra0;  *(v8bf*)&As[0][smm][skk + 8] = ra1;
  *(v8bf*)&Bs[0][smm][skk] = rb0;  *(v8bf*)&Bs[0][smm][skk + 8] = rb1;

  v8f acc00 = {}, acc01 = {}, acc10 = {}, acc11 = {};

  for (int it = 0; it < nIter; ++it) {
    __syncthreads();
    const int cur  = it & 1;
    const bool more = (it + 1) < nIter;

    if (more) {                       // fetch next tile into registers
      const int k1 = (it + 1) * 32;
      ra0 = *(const v8bf*)(aPtr + k1);     ra1 = *(const v8bf*)(aPtr + k1 + 8);
      rb0 = *(const v8bf*)(bPtr + k1);     rb1 = *(const v8bf*)(bPtr + k1 + 8);
    }
    if ((it + 2) < nIter) {           // L2 prefetch two tiles ahead
      const int k2 = (it + 2) * 32;
      __builtin_prefetch(aPtr + k2, 0, 1);
      __builtin_prefetch(bPtr + k2, 0, 1);
    }

    v16bf a0 = ldfrag(&As[cur][wm + mlane][khalf]);
    v16bf a1 = ldfrag(&As[cur][wm + 16 + mlane][khalf]);
    v16bf b0 = ldfrag(&Bs[cur][wn + mlane][khalf]);
    v16bf b1 = ldfrag(&Bs[cur][wn + 16 + mlane][khalf]);

    acc00 = __builtin_amdgcn_wmma_f32_16x16x32_bf16(false, a0, false, b0,
                                                    (short)0, acc00, false, false);
    acc01 = __builtin_amdgcn_wmma_f32_16x16x32_bf16(false, a0, false, b1,
                                                    (short)0, acc01, false, false);
    acc10 = __builtin_amdgcn_wmma_f32_16x16x32_bf16(false, a1, false, b0,
                                                    (short)0, acc10, false, false);
    acc11 = __builtin_amdgcn_wmma_f32_16x16x32_bf16(false, a1, false, b1,
                                                    (short)0, acc11, false, false);

    if (more) {                       // stage next tile into the other buffer
      const int nxt = 1 - cur;
      *(v8bf*)&As[nxt][smm][skk] = ra0;  *(v8bf*)&As[nxt][smm][skk + 8] = ra1;
      *(v8bf*)&Bs[nxt][smm][skk] = rb0;  *(v8bf*)&Bs[nxt][smm][skk + 8] = rb1;
    }
  }

  // Epilogue. C/D layout: VGPR r -> M = r (+8 for lanes 16-31); N = lane&15.
  const int mbase = m0blk + wm + ((lane >= 16) ? 8 : 0);
  const int nbase = n0blk + wn + mlane;
  #pragma unroll
  for (int r = 0; r < 8; ++r) {
    const int m0 = mbase + r;
    const int m1 = m0 + 16;
    if constexpr (EPI == 0) {
      __bf16* Cb = (__bf16*)C + cBatch * (size_t)b;
      Cb[(size_t)m0 * ldc + nbase]      = (__bf16)acc00[r];
      Cb[(size_t)m0 * ldc + nbase + 16] = (__bf16)acc01[r];
      Cb[(size_t)m1 * ldc + nbase]      = (__bf16)acc10[r];
      Cb[(size_t)m1 * ldc + nbase + 16] = (__bf16)acc11[r];
    } else {
      float* Cf = (float*)C + cBatch * (size_t)b;
      float v00 = acc00[r] + bias[nbase];
      float v01 = acc01[r] + bias[nbase + 16];
      float v10 = acc10[r] + bias[nbase];
      float v11 = acc11[r] + bias[nbase + 16];
      v00 = 0.5f * v00 * (1.0f + erff(v00 * 0.7071067811865475f));
      v01 = 0.5f * v01 * (1.0f + erff(v01 * 0.7071067811865475f));
      v10 = 0.5f * v10 * (1.0f + erff(v10 * 0.7071067811865475f));
      v11 = 0.5f * v11 * (1.0f + erff(v11 * 0.7071067811865475f));
      Cf[(size_t)m0 * ldc + nbase]      = v00;
      Cf[(size_t)m0 * ldc + nbase + 16] = v01;
      Cf[(size_t)m1 * ldc + nbase]      = v10;
      Cf[(size_t)m1 * ldc + nbase + 16] = v11;
    }
  }
}

// ---------------- x_qin = tanh(h @ Wpre^T + bpre) * pi ----------------
__global__ __launch_bounds__(256)
void preq_kernel(const float* __restrict__ h, const float* __restrict__ Wpre,
                 const float* __restrict__ bpre, float* __restrict__ xqin, int rows) {
  int i = blockIdx.x * blockDim.x + threadIdx.x;   // rows*8 threads
  if (i >= rows * 8) return;
  int r = i >> 3, q = i & 7;
  float s = bpre[q];
  const float* hr = h + (size_t)r * 256;
  const float* wq = Wpre + (size_t)q * 256;
  #pragma unroll 8
  for (int k = 0; k < 256; ++k) s += hr[k] * wq[k];
  xqin[i] = tanhf(s) * 3.14159265358979323846f;
}

// ---------------- 8-qubit state-vector simulator ----------------
// One wave32 per sample. 256 amplitudes: lane bits = qubit bits 7..3,
// per-lane slots k=0..7 = qubit bits 2..0 (global index g = (lane<<3)|k).
// Wire w maps to bit p = 7 - w (C-order flatten of (2,)*8).

template<int P>
__device__ __forceinline__ void ry_gate(float c, float s, float (&re)[8],
                                        float (&im)[8], int lane) {
  if constexpr (P < 3) {
    constexpr int m = 1 << P;
    #pragma unroll
    for (int k = 0; k < 8; ++k) {
      if ((k & m) == 0) {
        const int k1 = k | m;
        float r0 = re[k], i0 = im[k], r1 = re[k1], i1 = im[k1];
        re[k]  = c * r0 - s * r1;  im[k]  = c * i0 - s * i1;
        re[k1] = s * r0 + c * r1;  im[k1] = s * i0 + c * i1;
      }
    }
  } else {
    constexpr int lm = 1 << (P - 3);
    const float sg = ((lane >> (P - 3)) & 1) ? s : -s;
    #pragma unroll
    for (int k = 0; k < 8; ++k) {
      float orr = __shfl_xor(re[k], lm, 32);
      float oii = __shfl_xor(im[k], lm, 32);
      re[k] = c * re[k] + sg * orr;
      im[k] = c * im[k] + sg * oii;
    }
  }
}

template<int P>
__device__ __forceinline__ void rx_gate(float c, float s, float (&re)[8],
                                        float (&im)[8], int lane) {
  // RX = [[c, -is], [-is, c]] : symmetric, same update for both halves.
  if constexpr (P < 3) {
    constexpr int m = 1 << P;
    #pragma unroll
    for (int k = 0; k < 8; ++k) {
      if ((k & m) == 0) {
        const int k1 = k | m;
        float r0 = re[k], i0 = im[k], r1 = re[k1], i1 = im[k1];
        re[k]  = c * r0 + s * i1;  im[k]  = c * i0 - s * r1;
        re[k1] = c * r1 + s * i0;  im[k1] = c * i1 - s * r0;
      }
    }
  } else {
    constexpr int lm = 1 << (P - 3);
    #pragma unroll
    for (int k = 0; k < 8; ++k) {
      float orr = __shfl_xor(re[k], lm, 32);
      float oii = __shfl_xor(im[k], lm, 32);
      re[k] = c * re[k] + s * oii;
      im[k] = c * im[k] - s * orr;
    }
  }
}

template<int CP, int TP>
__device__ __forceinline__ void cnot_gate(float (&re)[8], float (&im)[8], int lane) {
  if constexpr (TP < 3) {
    constexpr int m = 1 << TP;
    #pragma unroll
    for (int k = 0; k < 8; ++k) {
      if ((k & m) == 0) {
        const int k1 = k | m;
        bool cb;
        if constexpr (CP < 3) cb = ((k >> CP) & 1) != 0;
        else                  cb = ((lane >> (CP - 3)) & 1) != 0;
        float r0 = re[k], i0 = im[k], r1 = re[k1], i1 = im[k1];
        re[k]  = cb ? r1 : r0;  im[k]  = cb ? i1 : i0;
        re[k1] = cb ? r0 : r1;  im[k1] = cb ? i0 : i1;
      }
    }
  } else {
    constexpr int lm = 1 << (TP - 3);
    #pragma unroll
    for (int k = 0; k < 8; ++k) {
      float orr = __shfl_xor(re[k], lm, 32);
      float oii = __shfl_xor(im[k], lm, 32);
      bool cb;
      if constexpr (CP < 3) cb = ((k >> CP) & 1) != 0;
      else                  cb = ((lane >> (CP - 3)) & 1) != 0;
      re[k] = cb ? orr : re[k];
      im[k] = cb ? oii : im[k];
    }
  }
}

__global__ __launch_bounds__(256)
void quantum_kernel(const float* __restrict__ xqin, const float* __restrict__ qwts,
                    float* __restrict__ qout, int nSamples) {
  const int lane   = threadIdx.x & 31;
  const int sample = blockIdx.x * (blockDim.x >> 5) + (threadIdx.x >> 5);
  if (sample >= nSamples) return;   // wave-uniform

  float re[8], im[8];
  #pragma unroll
  for (int k = 0; k < 8; ++k) { re[k] = 0.0f; im[k] = 0.0f; }
  if (lane == 0) re[0] = 1.0f;      // |0...0>

  float th[8];
  #pragma unroll
  for (int w = 0; w < 8; ++w) th[w] = xqin[(size_t)sample * 8 + w];
  float qv[16];
  #pragma unroll
  for (int i = 0; i < 16; ++i) qv[i] = qwts[i];

  // AngleEmbedding: RY(th[w]) on wire w (bit 7-w)
  { float t;
    t = 0.5f * th[0]; ry_gate<7>(cosf(t), sinf(t), re, im, lane);
    t = 0.5f * th[1]; ry_gate<6>(cosf(t), sinf(t), re, im, lane);
    t = 0.5f * th[2]; ry_gate<5>(cosf(t), sinf(t), re, im, lane);
    t = 0.5f * th[3]; ry_gate<4>(cosf(t), sinf(t), re, im, lane);
    t = 0.5f * th[4]; ry_gate<3>(cosf(t), sinf(t), re, im, lane);
    t = 0.5f * th[5]; ry_gate<2>(cosf(t), sinf(t), re, im, lane);
    t = 0.5f * th[6]; ry_gate<1>(cosf(t), sinf(t), re, im, lane);
    t = 0.5f * th[7]; ry_gate<0>(cosf(t), sinf(t), re, im, lane);
  }

  // BasicEntanglerLayers: L=2 layers of RX + ring CNOT
  for (int l = 0; l < 2; ++l) {
    const float* a = &qv[l * 8];
    float t;
    t = 0.5f * a[0]; rx_gate<7>(cosf(t), sinf(t), re, im, lane);
    t = 0.5f * a[1]; rx_gate<6>(cosf(t), sinf(t), re, im, lane);
    t = 0.5f * a[2]; rx_gate<5>(cosf(t), sinf(t), re, im, lane);
    t = 0.5f * a[3]; rx_gate<4>(cosf(t), sinf(t), re, im, lane);
    t = 0.5f * a[4]; rx_gate<3>(cosf(t), sinf(t), re, im, lane);
    t = 0.5f * a[5]; rx_gate<2>(cosf(t), sinf(t), re, im, lane);
    t = 0.5f * a[6]; rx_gate<1>(cosf(t), sinf(t), re, im, lane);
    t = 0.5f * a[7]; rx_gate<0>(cosf(t), sinf(t), re, im, lane);
    cnot_gate<7, 6>(re, im, lane);   // (wire0 -> wire1)
    cnot_gate<6, 5>(re, im, lane);
    cnot_gate<5, 4>(re, im, lane);
    cnot_gate<4, 3>(re, im, lane);
    cnot_gate<3, 2>(re, im, lane);
    cnot_gate<2, 1>(re, im, lane);
    cnot_gate<1, 0>(re, im, lane);
    cnot_gate<0, 7>(re, im, lane);   // (wire7 -> wire0)
  }

  // <Z_w> = sum_g (-1)^{bit_{7-w}(g)} |amp_g|^2
  float z[8];
  #pragma unroll
  for (int w = 0; w < 8; ++w) z[w] = 0.0f;
  #pragma unroll
  for (int k = 0; k < 8; ++k) {
    float pk = re[k] * re[k] + im[k] * im[k];
    int g = (lane << 3) | k;
    #pragma unroll
    for (int w = 0; w < 8; ++w) {
      int p = 7 - w;
      z[w] += ((g >> p) & 1) ? -pk : pk;
    }
  }
  #pragma unroll
  for (int off = 16; off > 0; off >>= 1) {
    #pragma unroll
    for (int w = 0; w < 8; ++w) z[w] += __shfl_xor(z[w], off, 32);
  }
  if (lane == 0) {
    #pragma unroll
    for (int w = 0; w < 8; ++w) qout[(size_t)sample * 8 + w] = z[w];
  }
}

// ---------------- out = qout @ Wpost^T + bpost ----------------
__global__ __launch_bounds__(256)
void post_kernel(const float* __restrict__ qout, const float* __restrict__ Wpost,
                 const float* __restrict__ bpost, float* __restrict__ out, int rows) {
  int i = blockIdx.x * blockDim.x + threadIdx.x;   // rows*256 threads
  if (i >= rows * 256) return;
  int r = i >> 8, hcol = i & 255;
  float s = bpost[hcol];
  const float* qr = qout + (size_t)r * 8;
  const float* wh = Wpost + (size_t)hcol * 8;
  #pragma unroll
  for (int q = 0; q < 8; ++q) s += qr[q] * wh[q];
  out[i] = s;
}

// ---------------------------------------------------------------------------
extern "C" void kernel_launch(void* const* d_in, const int* in_sizes, int n_in,
                              void* d_out, int out_size, void* d_ws, size_t ws_size,
                              hipStream_t stream) {
  (void)in_sizes; (void)n_in; (void)out_size; (void)ws_size;

  const float* x     = (const float*)d_in[0];   // (16,1024,256)
  const float* adj   = (const float*)d_in[1];   // (1024,1024)
  const float* Wg    = (const float*)d_in[2];   // (256,256)
  const float* bg    = (const float*)d_in[3];   // (256)
  const float* Wpre  = (const float*)d_in[4];   // (8,256)
  const float* bpre  = (const float*)d_in[5];   // (8)
  const float* qwts  = (const float*)d_in[6];   // (2,8)
  const float* Wpost = (const float*)d_in[7];   // (256,8)
  const float* bpost = (const float*)d_in[8];   // (256)
  float* out = (float*)d_out;

  constexpr int B = 16, N = 1024, H = 256, NQ = 8;
  constexpr int BN = B * N;                     // 16384

  char* ws = (char*)d_ws;
  size_t off = 0;
  auto carve = [&](size_t bytes) -> void* {
    void* p = ws + off;
    off = (off + bytes + 255) & ~(size_t)255;
    return p;
  };
  __bf16* xbt  = (__bf16*)carve((size_t)B * N * H * 2);   // 8 MB  (B, H, N) transposed
  __bf16* adjb = (__bf16*)carve((size_t)N * N * 2);       // 2 MB
  __bf16* wgb  = (__bf16*)carve((size_t)H * H * 2);       // 128 KB
  __bf16* aggb = (__bf16*)carve((size_t)B * N * H * 2);   // 8 MB
  float*  hbuf = (float*)carve((size_t)BN * H * 4);       // 16 MB
  float*  xqin = (float*)carve((size_t)BN * NQ * 4);      // 512 KB
  float*  qout = (float*)carve((size_t)BN * NQ * 4);      // 512 KB

  // 1) conversions: x -> transposed bf16 (B,H,N); adj, Wg -> bf16
  transpose_cvt_kernel<<<dim3(H / 32, N / 32, B), dim3(32, 8), 0, stream>>>(
      x, xbt, N, H);
  cvt_f32_bf16_kernel<<<(N * N + 255) / 256, 256, 0, stream>>>(adj, adjb, N * N);
  cvt_f32_bf16_kernel<<<(H * H + 255) / 256, 256, 0, stream>>>(Wg, wgb, H * H);

  // 2) agg[b] = adj @ x[b]   (M=1024, N=256, K=1024), Bt = xbt[b] (256x1024)
  gemm_wmma_bf16<0><<<dim3(H / 64, N / 64, B), 128, 0, stream>>>(
      adjb, 0, xbt, (size_t)N * H, aggb, (size_t)N * H, nullptr,
      /*K=*/N, /*lda=*/N, /*ldb=*/N, /*ldc=*/H);

  // 3) h = GELU(agg @ Wg^T + bg)   (M=16384, N=256, K=256), Bt = Wg (256x256)
  gemm_wmma_bf16<1><<<dim3(H / 64, BN / 64, 1), 128, 0, stream>>>(
      aggb, 0, wgb, 0, hbuf, 0, bg,
      /*K=*/H, /*lda=*/H, /*ldb=*/H, /*ldc=*/H);

  // 4) x_qin = tanh(h @ Wpre^T + bpre) * pi
  preq_kernel<<<(BN * NQ + 255) / 256, 256, 0, stream>>>(hbuf, Wpre, bpre, xqin, BN);

  // 5) 8-qubit circuit, one wave32 per sample
  quantum_kernel<<<BN / 8, 256, 0, stream>>>(xqin, qwts, qout, BN);

  // 6) out = qout @ Wpost^T + bpost
  post_kernel<<<(BN * H + 255) / 256, 256, 0, stream>>>(qout, Wpost, bpost, out, BN);
}
